// VISBlock_7344394076838
// MI455X (gfx1250) — matrix operand order repaired and checked by reading, not compiled, not run on hardware
//
#include <hip/hip_runtime.h>
#include <hip/hip_bf16.h>
#include <math.h>

// ---------------------------------------------------------------------------
// Types for WMMA (gfx1250, wave32)
// ---------------------------------------------------------------------------
typedef __attribute__((ext_vector_type(16))) __bf16 v16bf;
typedef __attribute__((ext_vector_type(8)))  __bf16 v8bf;
typedef __attribute__((ext_vector_type(4)))  __bf16 v4bf;
typedef __attribute__((ext_vector_type(8)))  float  v8f;

#if defined(__gfx1250__) && __has_builtin(__builtin_amdgcn_global_load_async_to_lds_b128)
#define USE_ASYNC_LDS 1
#else
#define USE_ASYNC_LDS 0
#endif

#if USE_ASYNC_LDS
#define AS1 __attribute__((address_space(1)))
#define AS3 __attribute__((address_space(3)))
typedef int v4i_vs __attribute__((vector_size(16)));  // matches builtin's V4i
__device__ __forceinline__ void async_b128(const __bf16* g, __bf16* l) {
  __builtin_amdgcn_global_load_async_to_lds_b128(
      (AS1 v4i_vs*)(g), (AS3 v4i_vs*)(l), 0, 0);
}
__device__ __forceinline__ void wait_async0() {
#if __has_builtin(__builtin_amdgcn_s_wait_asynccnt)
  __builtin_amdgcn_s_wait_asynccnt(0);
#else
  asm volatile("s_wait_asynccnt 0x0" ::: "memory");
#endif
}
#endif

// ---------------------------------------------------------------------------
// fp32 -> bf16 conversion (vectorized x4)
// ---------------------------------------------------------------------------
__global__ __launch_bounds__(256)
void cvt_bf16_kernel(const float* __restrict__ x, __bf16* __restrict__ y, int n4)
{
  const int i = blockIdx.x * 256 + threadIdx.x;
  if (i < n4) {
    const float4 f = reinterpret_cast<const float4*>(x)[i];
    v4bf r;
    r[0] = (__bf16)f.x; r[1] = (__bf16)f.y; r[2] = (__bf16)f.z; r[3] = (__bf16)f.w;
    reinterpret_cast<v4bf*>(y)[i] = r;
  }
}

// ---------------------------------------------------------------------------
// Tiled GEMM: out[M,N] = act(A[M,K] @ W[N,K]^T + bias) (+ resid)
// A, W bf16 in memory. Double-buffered LDS stages filled by async
// load-to-LDS (ASYNCcnt) overlapping WMMA compute; one barrier per 64-wide
// K stage (8 WMMAs). Block tile 64x128, 8 waves (4x2), each wave 16x64.
// Requires M%64==0, N%128==0, K%64==0.
// ---------------------------------------------------------------------------
#define TM 64
#define TN 128
#define TKS 64      // K elems per stage (2 WMMA K-steps)
#define LDS_K 72    // bf16 elems per LDS row (64 + 8 pad, keeps 16B alignment)

__global__ __launch_bounds__(256)
void gemm_wmma_bf16(const __bf16* __restrict__ A, const __bf16* __restrict__ W,
                    const float* __restrict__ bias, const float* __restrict__ resid,
                    float* __restrict__ outf, __bf16* __restrict__ outb,
                    int M, int N, int K, int act)
{
  __shared__ __bf16 As[2][TM * LDS_K];
  __shared__ __bf16 Bs[2][TN * LDS_K];
  const int tid  = threadIdx.x;
  const int lane = tid & 31;
  const int wave = tid >> 5;
  const int wrow = wave & 3;   // 4 wave-rows * 16 = 64
  const int wcol = wave >> 2;  // 2 wave-cols * 64 = 128
  const int mBase = blockIdx.y * TM;
  const int nBase = blockIdx.x * TN;

  v8f acc[4];
  for (int t = 0; t < 4; ++t)
    for (int j = 0; j < 8; ++j) acc[t][j] = 0.0f;

  // A stage: 64 rows x 64 bf16 -> 2 b128 chunks per thread
  const int aRow = tid >> 2;            // 0..63
  const int aCol = (tid & 3) * 8;       // 0,8,16,24 (+32 second chunk)
  // W stage: 128 rows x 64 bf16 -> 4 b128 chunks per thread
  const int bRow = tid >> 1;            // 0..127
  const int bCol = (tid & 1) * 32;      // 0,32 (covers 4 chunks of 8)

  const __bf16* ga = A + (size_t)(mBase + aRow) * K + aCol;
  const __bf16* gw = W + (size_t)(nBase + bRow) * K + bCol;

  auto stage = [&](int buf, int k0) {
    __bf16* la = &As[buf][aRow * LDS_K + aCol];
    __bf16* lb = &Bs[buf][bRow * LDS_K + bCol];
    const __bf16* sa = ga + k0;
    const __bf16* sw = gw + k0;
#if USE_ASYNC_LDS
    async_b128(sa, la);
    async_b128(sa + 32, la + 32);
    async_b128(sw, lb);
    async_b128(sw + 8, lb + 8);
    async_b128(sw + 16, lb + 16);
    async_b128(sw + 24, lb + 24);
#else
    *reinterpret_cast<uint4*>(la)      = *reinterpret_cast<const uint4*>(sa);
    *reinterpret_cast<uint4*>(la + 32) = *reinterpret_cast<const uint4*>(sa + 32);
#pragma unroll
    for (int c = 0; c < 4; ++c)
      *reinterpret_cast<uint4*>(lb + 8 * c) =
          *reinterpret_cast<const uint4*>(sw + 8 * c);
#endif
  };

  const int nk = K / TKS;
  stage(0, 0);

  const int mrow = lane & 15;
  const int kb  = (lane < 16) ? 0 : 16;    // A frag K select
  const int kb2 = (lane >> 4) << 4;        // B frag K select

  for (int kt = 0; kt < nk; ++kt) {
    const int cur = kt & 1;
#if USE_ASYNC_LDS
    wait_async0();            // only this stage's fills are outstanding here
#endif
    __syncthreads();          // all waves' fills visible; prev reads finished
    if (kt + 1 < nk) stage(cur ^ 1, (kt + 1) * TKS);   // overlaps compute
    if (kt + 2 < nk) {
      __builtin_prefetch(ga + (kt + 2) * TKS, 0, 1);
      __builtin_prefetch(gw + (kt + 2) * TKS, 0, 1);
    }

    const __bf16* asrc = &As[cur][(wrow * 16 + mrow) * LDS_K];
#pragma unroll
    for (int s = 0; s < 2; ++s) {
      const int ko = s * 32;
      // A fragment 16x32: lanes 0-15 K=0..15; lanes 16-31 K=16..31
      v8bf alo = *reinterpret_cast<const v8bf*>(asrc + ko + kb);
      v8bf ahi = *reinterpret_cast<const v8bf*>(asrc + ko + kb + 8);
      v16bf af = __builtin_shufflevector(alo, ahi,
          0,1,2,3,4,5,6,7,8,9,10,11,12,13,14,15);
      // B fragment 32x16: n = lane&15; lanes 0-15 K=0..15, lanes 16-31 K=16..31
#pragma unroll
      for (int nt = 0; nt < 4; ++nt) {
        const __bf16* br =
            &Bs[cur][(wcol * 64 + nt * 16 + (lane & 15)) * LDS_K + ko + kb2];
        v16bf bfm = *reinterpret_cast<const v16bf*>(br);
        acc[nt] = __builtin_amdgcn_wmma_f32_16x16x32_bf16(
            false, af, false, bfm, (short)0, acc[nt], false, false);
      }
    }
  }

  // Epilogue. C layout: VGPR i -> (M=i, N=lane) lanes 0-15; (M=8+i, N=lane-16)
  const int nOff = lane & 15;
  const int rSel = (lane >> 4) * 8;
#pragma unroll
  for (int nt = 0; nt < 4; ++nt) {
    const int gn = nBase + wcol * 64 + nt * 16 + nOff;
    const float bb = bias ? bias[gn] : 0.0f;
    float v[8];
#pragma unroll
    for (int i = 0; i < 8; ++i) {
      v[i] = acc[nt][i] + bb;
      if (act == 1) v[i] = 0.5f * v[i] * (1.0f + erff(v[i] * 0.70710678118654752f));
    }
    const size_t rb = (size_t)(mBase + wrow * 16 + rSel) * N + gn;
    if (outb) {
#pragma unroll
      for (int i = 0; i < 8; ++i) outb[rb + (size_t)i * N] = (__bf16)v[i];
    } else if (resid) {
#pragma unroll
      for (int i = 0; i < 8; ++i) {
        const size_t idx = rb + (size_t)i * N;
        outf[idx] = v[i] + resid[idx];
      }
    } else {
#pragma unroll
      for (int i = 0; i < 8; ++i) outf[rb + (size_t)i * N] = v[i];
    }
  }
}

// ---------------------------------------------------------------------------
// LayerNorm over last dim (C), one row per workgroup, bf16 output
// ---------------------------------------------------------------------------
__global__ __launch_bounds__(256)
void layernorm_kernel(const float* __restrict__ x, const float* __restrict__ g,
                      const float* __restrict__ b, __bf16* __restrict__ out, int C)
{
  __shared__ float red[256];
  const int row = blockIdx.x, tid = threadIdx.x;
  const float* xp = x + (size_t)row * C;
  float s = 0.0f;
  for (int c = tid; c < C; c += 256) s += xp[c];
  red[tid] = s; __syncthreads();
  for (int st = 128; st > 0; st >>= 1) { if (tid < st) red[tid] += red[tid + st]; __syncthreads(); }
  const float mean = red[0] / (float)C; __syncthreads();
  float s2 = 0.0f;
  for (int c = tid; c < C; c += 256) { float d = xp[c] - mean; s2 += d * d; }
  red[tid] = s2; __syncthreads();
  for (int st = 128; st > 0; st >>= 1) { if (tid < st) red[tid] += red[tid + st]; __syncthreads(); }
  const float rstd = rsqrtf(red[0] / (float)C + 1e-5f);
  __bf16* op = out + (size_t)row * C;
  for (int c = tid; c < C; c += 256)
    op[c] = (__bf16)((xp[c] - mean) * rstd * g[c] + b[c]);
}

// ---------------------------------------------------------------------------
// L2-normalize rows in place, x[rows, C]
// ---------------------------------------------------------------------------
__global__ __launch_bounds__(256)
void rownorm_kernel(float* __restrict__ x, int C)
{
  __shared__ float red[256];
  const int row = blockIdx.x, tid = threadIdx.x;
  float* xp = x + (size_t)row * C;
  float s = 0.0f;
  for (int c = tid; c < C; c += 256) { float v = xp[c]; s += v * v; }
  red[tid] = s; __syncthreads();
  for (int st = 128; st > 0; st >>= 1) { if (tid < st) red[tid] += red[tid + st]; __syncthreads(); }
  const float inv = 1.0f / fmaxf(sqrtf(red[0]), 1e-12f);
  __syncthreads();
  for (int c = tid; c < C; c += 256) xp[c] *= inv;
}

// ---------------------------------------------------------------------------
// sim[b,l,n] = dot(l[b,l,:256], vis[b,n,:256]); grid (L, B)
// ---------------------------------------------------------------------------
__global__ __launch_bounds__(256)
void simdot_kernel(const float* __restrict__ lq, const float* __restrict__ vis,
                   float* __restrict__ sim, int L, int N)
{
  __shared__ float lr[256];
  const int l = blockIdx.x, b = blockIdx.y, tid = threadIdx.x;
  lr[tid] = lq[(size_t)(b * L + l) * 256 + tid];
  __syncthreads();
  for (int n = tid; n < N; n += 256) {
    const float* vp = vis + (size_t)(b * N + n) * 256;
    float d = 0.0f;
    for (int c = 0; c < 256; ++c) d += lr[c] * vp[c];
    sim[(size_t)(b * L + l) * N + n] = d;
  }
}

// ---------------------------------------------------------------------------
// Exact top-k mask via 32-bit radix select; ties by lowest index (lax.top_k)
// ---------------------------------------------------------------------------
__global__ __launch_bounds__(256)
void topk_kernel(const float* __restrict__ sim, float* __restrict__ tmask,
                 int L, int N, int k)
{
  __shared__ unsigned u[1024];
  __shared__ int cnt[256];
  const int l = blockIdx.x, b = blockIdx.y, tid = threadIdx.x;
  const float* sp = sim + (size_t)(b * L + l) * N;
  for (int n = tid; n < N; n += 256) {
    unsigned bits = __float_as_uint(sp[n]);
    u[n] = bits ^ ((bits & 0x80000000u) ? 0xFFFFFFFFu : 0x80000000u);
  }
  __syncthreads();
  unsigned prefix = 0; int kr = k;
  for (int bit = 31; bit >= 0; --bit) {
    const unsigned hi = (prefix >> bit) | 1u;
    int c = 0;
    for (int n = tid; n < N; n += 256) c += ((u[n] >> bit) == hi) ? 1 : 0;
    cnt[tid] = c; __syncthreads();
    for (int st = 128; st > 0; st >>= 1) { if (tid < st) cnt[tid] += cnt[tid + st]; __syncthreads(); }
    const int total = cnt[0]; __syncthreads();
    if (total >= kr) prefix |= (1u << bit); else kr -= total;
  }
  const unsigned T = prefix;
  float* mp = tmask + (size_t)(b * L + l) * N;
  for (int n = tid; n < N; n += 256) {
    float mv = 0.0f;
    if (u[n] > T) mv = 1.0f;
    else if (u[n] == T) {
      int rank = 0;
      for (int j = 0; j < n; ++j) rank += (u[j] == T) ? 1 : 0;
      if (rank < kr) mv = 1.0f;
    }
    mp[n] = mv;
  }
}

// ---------------------------------------------------------------------------
// vq[b,l,c] = sum_n mask*softmax(sim+mask2)[n] * v_feats[b,n,c] / k
// ---------------------------------------------------------------------------
__global__ __launch_bounds__(256)
void pool_kernel(const float* __restrict__ sim, const float* __restrict__ tmask,
                 const float* __restrict__ vf, float* __restrict__ vq,
                 int L, int N, int C, int k)
{
  __shared__ float p[1024];
  __shared__ int nidx[1024];
  __shared__ float red[256];
  __shared__ int ncnt;
  const int l = blockIdx.x, b = blockIdx.y, tid = threadIdx.x;
  if (tid == 0) ncnt = 0;
  const float* sp = sim + (size_t)(b * L + l) * N;
  const float* mp = tmask + (size_t)(b * L + l) * N;
  float mx = -1e30f;
  for (int n = tid; n < N; n += 256) {
    const float v = (mp[n] > 0.5f) ? sp[n] : -1e30f;
    p[n] = v;
    mx = fmaxf(mx, v);
  }
  red[tid] = mx; __syncthreads();
  for (int st = 128; st > 0; st >>= 1) { if (tid < st) red[tid] = fmaxf(red[tid], red[tid + st]); __syncthreads(); }
  mx = red[0]; __syncthreads();
  float s = 0.0f;
  for (int n = tid; n < N; n += 256) {
    const float e = (p[n] > -1e29f) ? expf(p[n] - mx) : 0.0f;
    p[n] = e; s += e;
  }
  red[tid] = s; __syncthreads();
  for (int st = 128; st > 0; st >>= 1) { if (tid < st) red[tid] += red[tid + st]; __syncthreads(); }
  const float invZ = 1.0f / red[0]; __syncthreads();
  for (int n = tid; n < N; n += 256) {
    p[n] *= invZ;
    if (mp[n] > 0.5f) { int pos = atomicAdd(&ncnt, 1); nidx[pos] = n; }
  }
  __syncthreads();
  const int cn = ncnt;
  const float invk = 1.0f / (float)k;
  for (int c = tid; c < C; c += 256) {
    float acc = 0.0f;
    for (int t = 0; t < cn; ++t) {
      const int n = nidx[t];
      acc += p[n] * vf[(size_t)(b * N + n) * C + c];
    }
    vq[(size_t)(b * L + l) * C + c] = acc * invk;
  }
}

// ---------------------------------------------------------------------------
// Attention, Nq=20 queries. grid (L, HEADS, B). kv[b,n,1024]: k=[0:512),
// v=[512:1024), head h at h*64. Output bf16 (feeds the proj GEMM).
// ---------------------------------------------------------------------------
__global__ __launch_bounds__(256)
void attn_kernel(const float* __restrict__ q, const float* __restrict__ kv,
                 const float* __restrict__ tmask, __bf16* __restrict__ o,
                 int L, int Nk, float scale)
{
  __shared__ float qs[64];
  __shared__ float s[1024];
  __shared__ float red[256];
  const int l = blockIdx.x, h = blockIdx.y, b = blockIdx.z, tid = threadIdx.x;
  const float* qp = q + (size_t)(b * L + l) * 512 + h * 64;
  if (tid < 64) qs[tid] = qp[tid];
  __syncthreads();
  for (int n = tid; n < Nk; n += 256) {
    const float* kp = kv + (size_t)(b * Nk + n) * 1024 + h * 64;
    float d = 0.0f;
    for (int c = 0; c < 64; ++c) d += qs[c] * kp[c];
    d *= scale;
    if (tmask) d += tmask[(size_t)(b * L + l) * 1024 + n] * 10000.0f - 10000.0f;
    s[n] = d;
  }
  __syncthreads();
  float m = -1e30f;
  for (int n = tid; n < Nk; n += 256) m = fmaxf(m, s[n]);
  red[tid] = m; __syncthreads();
  for (int st = 128; st > 0; st >>= 1) { if (tid < st) red[tid] = fmaxf(red[tid], red[tid + st]); __syncthreads(); }
  m = red[0]; __syncthreads();
  float ps = 0.0f;
  for (int n = tid; n < Nk; n += 256) { const float e = expf(s[n] - m); s[n] = e; ps += e; }
  red[tid] = ps; __syncthreads();
  for (int st = 128; st > 0; st >>= 1) { if (tid < st) red[tid] += red[tid + st]; __syncthreads(); }
  const float Z = red[0]; __syncthreads();
  const int d = tid & 63, grp = tid >> 6;
  float acc = 0.0f;
  for (int n = grp; n < Nk; n += 4) {
    const float* vp = kv + (size_t)(b * Nk + n) * 1024 + 512 + h * 64;
    acc += s[n] * vp[d];
  }
  red[tid] = acc; __syncthreads();
  if (tid < 64) {
    const float r = red[d] + red[64 + d] + red[128 + d] + red[192 + d];
    o[(size_t)(b * L + l) * 512 + h * 64 + d] = (__bf16)(r / Z);
  }
}

// ---------------------------------------------------------------------------
// Host-side launch
// ---------------------------------------------------------------------------
extern "C" void kernel_launch(void* const* d_in, const int* in_sizes, int n_in,
                              void* d_out, int out_size, void* d_ws, size_t ws_size,
                              hipStream_t stream)
{
  (void)in_sizes; (void)n_in; (void)out_size; (void)ws_size;
  const int B = 16, N = 1024, L = 20, C = 768, HID = 512, HEADS = 8, MLPH = 3072, NB = 8;
  const int KTOP = 307;                 // int(1024 * 0.3)
  const int BL = B * L;                 // 320 (multiple of 64)
  const int BN = B * N;                 // 16384
  const float SCALE = 0.125f;           // 64^-0.5

  const float* v_feats = (const float*)d_in[0];
  const float* l_feats = (const float*)d_in[1];
  const float* ln_g    = (const float*)d_in[3];
  const float* ln_b    = (const float*)d_in[4];
  const float* Wq      = (const float*)d_in[5];
  const float* Wkv     = (const float*)d_in[6];
  const float* Wproj   = (const float*)d_in[7];
  const float* bproj   = (const float*)d_in[8];
  const float* W1      = (const float*)d_in[9];
  const float* b1      = (const float*)d_in[10];
  const float* W2      = (const float*)d_in[11];
  const float* b2      = (const float*)d_in[12];
  const float* Wl      = (const float*)d_in[13];
  const float* bl      = (const float*)d_in[14];
  const float* Wv      = (const float*)d_in[15];
  const float* bv      = (const float*)d_in[16];

  // workspace carve-up
  char* base = (char*)d_ws;
  size_t off = 0;
  auto carveF = [&](size_t n) -> float* {
    float* p = (float*)(base + off);
    off += ((n * sizeof(float)) + 255) & ~(size_t)255;
    return p;
  };
  auto carveB = [&](size_t n) -> __bf16* {
    __bf16* p = (__bf16*)(base + off);
    off += ((n * sizeof(__bf16)) + 255) & ~(size_t)255;
    return p;
  };
  // fp32 buffers
  float* kvb  = carveF((size_t)BN * 1024);  // KV projections
  float* vis  = carveF((size_t)BN * 256);
  float* lb   = carveF((size_t)BL * 256);
  float* sim  = carveF((size_t)BL * N);
  float* tmsk = carveF((size_t)BL * N);
  float* vq   = carveF((size_t)BL * C);
  float* qb   = carveF((size_t)BL * HID);
  // bf16 buffers
  __bf16* vf_bf   = carveB((size_t)BN * C);
  __bf16* lf_bf   = carveB((size_t)BL * C);
  __bf16* lnkv_bf = carveB((size_t)BN * C);
  __bf16* lnq_bf  = carveB((size_t)BL * C);
  __bf16* yb_bf   = carveB((size_t)BL * C);
  __bf16* ob_bf   = carveB((size_t)BL * HID);
  __bf16* hb_bf   = carveB((size_t)BL * MLPH);
  __bf16* wl_bf   = carveB((size_t)256 * C);
  __bf16* wv_bf   = carveB((size_t)256 * C);
  __bf16* wq_bf   = carveB((size_t)HID * C);
  __bf16* wkv_bf  = carveB((size_t)1024 * C);
  __bf16* wp_bf   = carveB((size_t)C * HID);
  __bf16* w1_bf   = carveB((size_t)MLPH * C);
  __bf16* w2_bf   = carveB((size_t)C * MLPH);

  auto cvt = [&](const float* src, __bf16* dst, size_t n) {
    const int n4 = (int)(n / 4);
    cvt_bf16_kernel<<<(n4 + 255) / 256, 256, 0, stream>>>(src, dst, n4);
  };
  auto gemm = [&](const __bf16* A, const __bf16* W, const float* bias,
                  const float* resid, float* outf, __bf16* outb,
                  int M, int Nn, int K, int act) {
    dim3 g(Nn / TN, M / TM);
    gemm_wmma_bf16<<<g, 256, 0, stream>>>(A, W, bias, resid, outf, outb,
                                          M, Nn, K, act);
  };

  // ---- similarity path ----
  cvt(v_feats, vf_bf, (size_t)BN * C);
  cvt(l_feats, lf_bf, (size_t)BL * C);
  cvt(Wv, wv_bf, (size_t)256 * C);
  cvt(Wl, wl_bf, (size_t)256 * C);
  gemm(vf_bf, wv_bf, bv, nullptr, vis, nullptr, BN, 256, C, 0);
  rownorm_kernel<<<BN, 256, 0, stream>>>(vis, 256);
  gemm(lf_bf, wl_bf, bl, nullptr, lb, nullptr, BL, 256, C, 0);
  rownorm_kernel<<<BL, 256, 0, stream>>>(lb, 256);
  simdot_kernel<<<dim3(L, B), 256, 0, stream>>>(lb, vis, sim, L, N);
  topk_kernel<<<dim3(L, B), 256, 0, stream>>>(sim, tmsk, L, N, KTOP);
  pool_kernel<<<dim3(L, B), 256, 0, stream>>>(sim, tmsk, v_feats, vq, L, N, C, KTOP);

  // ---- 8 transformer blocks ----
  for (int i = 0; i < NB; ++i) {
    const bool cross = (i < 4);
    const float* kvsrc = cross ? v_feats : vq;
    const int Mkv = cross ? BN : BL;
    const int Nk  = cross ? N : L;

    cvt(Wq + (size_t)i * HID * C, wq_bf, (size_t)HID * C);
    cvt(Wkv + (size_t)i * 1024 * C, wkv_bf, (size_t)1024 * C);
    cvt(Wproj + (size_t)i * C * HID, wp_bf, (size_t)C * HID);
    cvt(W1 + (size_t)i * MLPH * C, w1_bf, (size_t)MLPH * C);
    cvt(W2 + (size_t)i * C * MLPH, w2_bf, (size_t)C * MLPH);

    layernorm_kernel<<<BL, 256, 0, stream>>>(vq, ln_g + (i * 3 + 0) * C,
                                             ln_b + (i * 3 + 0) * C, lnq_bf, C);
    layernorm_kernel<<<Mkv, 256, 0, stream>>>(kvsrc, ln_g + (i * 3 + 1) * C,
                                              ln_b + (i * 3 + 1) * C, lnkv_bf, C);
    gemm(lnq_bf, wq_bf, nullptr, nullptr, qb, nullptr, BL, HID, C, 0);
    gemm(lnkv_bf, wkv_bf, nullptr, nullptr, kvb, nullptr, Mkv, 1024, C, 0);
    attn_kernel<<<dim3(L, HEADS, B), 256, 0, stream>>>(
        qb, kvb, cross ? tmsk : nullptr, ob_bf, L, Nk, SCALE);
    gemm(ob_bf, wp_bf, bproj + (size_t)i * C, vq, vq, nullptr, BL, C, HID, 0);
    layernorm_kernel<<<BL, 256, 0, stream>>>(vq, ln_g + (i * 3 + 2) * C,
                                             ln_b + (i * 3 + 2) * C, yb_bf, C);
    gemm(yb_bf, w1_bf, b1 + (size_t)i * MLPH, nullptr, nullptr, hb_bf,
         BL, MLPH, C, 1 /*gelu*/);
    gemm(hb_bf, w2_bf, b2 + (size_t)i * C, vq, vq, nullptr, BL, C, MLPH, 0);
  }

  (void)hipMemcpyAsync(d_out, vq, (size_t)BL * C * sizeof(float),
                       hipMemcpyDeviceToDevice, stream);
}
